// GATEdgeClassifier_45741401703145
// MI455X (gfx1250) — matrix operand order repaired and checked by reading, not compiled, not run on hardware
//
#include <hip/hip_runtime.h>

#define NN 10000
#define EE 320000

typedef __attribute__((ext_vector_type(16))) _Float16 v16h;
typedef __attribute__((ext_vector_type(8)))  float    v8f;

// ---------------- helpers ----------------

__device__ __forceinline__ unsigned f2key(float f) {
  unsigned u = __float_as_uint(f);
  return (u & 0x80000000u) ? ~u : (u | 0x80000000u);
}
__device__ __forceinline__ float key2f(unsigned k) {
  return (k & 0x80000000u) ? __uint_as_float(k ^ 0x80000000u) : __uint_as_float(~k);
}

__device__ __forceinline__ v16h pack16(float4 a0, float4 a1, float4 a2, float4 a3) {
  v16h r;
  r[0]=(_Float16)a0.x; r[1]=(_Float16)a0.y; r[2]=(_Float16)a0.z; r[3]=(_Float16)a0.w;
  r[4]=(_Float16)a1.x; r[5]=(_Float16)a1.y; r[6]=(_Float16)a1.z; r[7]=(_Float16)a1.w;
  r[8]=(_Float16)a2.x; r[9]=(_Float16)a2.y; r[10]=(_Float16)a2.z; r[11]=(_Float16)a2.w;
  r[12]=(_Float16)a3.x; r[13]=(_Float16)a3.y; r[14]=(_Float16)a3.z; r[15]=(_Float16)a3.w;
  return r;
}

// ---------------- generic kernels ----------------

__global__ void k_zero(float* __restrict__ p, long n) {
  long i = (long)blockIdx.x * blockDim.x + threadIdx.x;
  long st = (long)gridDim.x * blockDim.x;
  for (; i < n; i += st) p[i] = 0.0f;
}

// Pack f32 weight [K x Nc] into WMMA B-fragment order (f16), tile = 32(K) x 16(N),
// 512 halfs per tile laid out as [lane][16 contiguous halfs] so the GEMM reads one v16h.
__global__ void k_pack_b(const float* __restrict__ wsrc, _Float16* __restrict__ pb,
                         int K, int Nc) {
  int tid = blockIdx.x * blockDim.x + threadIdx.x;
  int total = K * Nc;
  if (tid >= total) return;
  int tile = tid >> 9;
  int within = tid & 511;
  int lane = within >> 4;
  int j = within & 15;
  int chunks_n = Nc >> 4;
  int kc = tile / chunks_n;
  int nt = tile - kc * chunks_n;
  int half = lane >> 4;
  int l = lane & 15;
  int kk = (kc << 5) + (half << 3) + j + ((j >> 3) << 3);   // B layout: low lanes K 0..7,16..23, high lanes K 8..15,24..31
  int col = (nt << 4) + l;
  pb[tid] = (_Float16)wsrc[(size_t)kk * Nc + col];
}

// C[M x Nc] = act(A[M x K](f32) @ PB(packed f16) + bias). M%16==0, K%32==0, Nc%64==0.
// One wave per block; each wave owns a 16-row x 64-col slab. Software-pipelined:
// next chunk's A/B loads are issued before the current chunk's 4 WMMAs.
__global__ __launch_bounds__(32) void k_gemm(const float* __restrict__ A,
    const _Float16* __restrict__ PB, const float* __restrict__ bias,
    float* __restrict__ Cout, int M, int K, int Nc, int act) {
  const int lane = threadIdx.x & 31;
  const int half = lane >> 4;
  const int l = lane & 15;
  const int m0 = blockIdx.x << 4;
  const int n0 = blockIdx.y << 6;
  const float* arow = A + (size_t)(m0 + l) * K;
  const int klo_off = half ? 8 : 0;     // A layout per ISA 7.12.2
  const int khi_off = half ? 24 : 16;
  const _Float16* bbase = PB + ((size_t)(n0 >> 4) << 9) + (lane << 4);
  const size_t bstride = ((size_t)(Nc >> 4)) << 9;

  v8f acc[4] = {{}, {}, {}, {}};

  // prologue: chunk 0
  float4 p0 = *(const float4*)(arow + klo_off);
  float4 p1 = *(const float4*)(arow + klo_off + 4);
  float4 p2 = *(const float4*)(arow + khi_off);
  float4 p3 = *(const float4*)(arow + khi_off + 4);
  v16h c0 = *(const v16h*)(bbase);
  v16h c1 = *(const v16h*)(bbase + 512);
  v16h c2 = *(const v16h*)(bbase + 1024);
  v16h c3 = *(const v16h*)(bbase + 1536);
  v16h af = pack16(p0, p1, p2, p3);

  for (int k0 = 0; k0 < K; k0 += 32) {
    const int kn = (k0 + 32 < K) ? (k0 + 32) : k0;   // clamp on last iter
    const float* an = arow + kn;
    const _Float16* bn = bbase + (size_t)(kn >> 5) * bstride;
    float4 q0 = *(const float4*)(an + klo_off);
    float4 q1 = *(const float4*)(an + klo_off + 4);
    float4 q2 = *(const float4*)(an + khi_off);
    float4 q3 = *(const float4*)(an + khi_off + 4);
    v16h d0 = *(const v16h*)(bn);
    v16h d1 = *(const v16h*)(bn + 512);
    v16h d2 = *(const v16h*)(bn + 1024);
    v16h d3 = *(const v16h*)(bn + 1536);
    acc[0] = __builtin_amdgcn_wmma_f32_16x16x32_f16(false, af, false, c0, (short)0, acc[0], false, false);
    acc[1] = __builtin_amdgcn_wmma_f32_16x16x32_f16(false, af, false, c1, (short)0, acc[1], false, false);
    acc[2] = __builtin_amdgcn_wmma_f32_16x16x32_f16(false, af, false, c2, (short)0, acc[2], false, false);
    acc[3] = __builtin_amdgcn_wmma_f32_16x16x32_f16(false, af, false, c3, (short)0, acc[3], false, false);
    af = pack16(q0, q1, q2, q3);
    c0 = d0; c1 = d1; c2 = d2; c3 = d3;
  }

#pragma unroll
  for (int t = 0; t < 4; ++t) {
    const int col = n0 + (t << 4) + l;
    float bb = bias ? bias[col] : 0.0f;
#pragma unroll
    for (int r = 0; r < 8; ++r) {
      float v = acc[t][r] + bb;
      if (act == 1) v = fmaxf(v, 0.0f);
      else if (act == 2) v = (v > 0.0f) ? v : (__expf(v) - 1.0f);
      Cout[(size_t)(m0 + r + (half << 3)) * Nc + col] = v;
    }
  }
}

// ---------------- encoder / graph prep ----------------

__global__ void k_enc1(const float* __restrict__ x, const float* __restrict__ w1,
                       const float* __restrict__ b1, float* __restrict__ henc) {
  int tid = blockIdx.x * blockDim.x + threadIdx.x;
  if (tid >= NN * 128) return;
  int n = tid >> 7, c = tid & 127;
  float v = x[n * 2] * w1[c] + x[n * 2 + 1] * w1[128 + c] + b1[c];
  henc[tid] = fmaxf(v, 0.0f);
}

__global__ void k_deg(const int* __restrict__ dst, const float* __restrict__ eattr,
                      float* __restrict__ deg, float* __restrict__ macc) {
  int e = blockIdx.x * blockDim.x + threadIdx.x;
  if (e >= EE) return;
  int d = dst[e];
  atomicAdd(deg + d, 1.0f);
  atomicAdd(macc + d * 3 + 0, eattr[e * 3 + 0]);
  atomicAdd(macc + d * 3 + 1, eattr[e * 3 + 1]);
  atomicAdd(macc + d * 3 + 2, eattr[e * 3 + 2]);
}

__global__ void k_meandiv(const float* __restrict__ macc, const float* __restrict__ deg,
                          float* __restrict__ meane) {
  int tid = blockIdx.x * blockDim.x + threadIdx.x;
  if (tid >= NN * 3) return;
  meane[tid] = macc[tid] / fmaxf(deg[tid / 3], 1.0f);
}

// ---------------- attention ----------------

__global__ void k_attnode(const float* __restrict__ xl, const float* __restrict__ as_,
                          const float* __restrict__ ad_, float* __restrict__ asrc,
                          float* __restrict__ adst, int H) {
  int tid = blockIdx.x * blockDim.x + threadIdx.x;
  if (tid >= NN * H) return;
  int n = tid / H, h = tid - n * H;
  const float* xr = xl + (size_t)n * H * 64 + h * 64;
  const float* sp = as_ + h * 64;
  const float* dp = ad_ + h * 64;
  float ss = 0.0f, dd = 0.0f;
#pragma unroll
  for (int c = 0; c < 64; ++c) { float v = xr[c]; ss += v * sp[c]; dd += v * dp[c]; }
  asrc[tid] = ss;
  adst[tid] = dd;
}

__global__ void k_wered(const float* __restrict__ We, const float* __restrict__ ae,
                        float* __restrict__ wr, int H) {
  int tid = threadIdx.x;
  if (tid >= 3 * H) return;
  int d3 = tid / H, h = tid - d3 * H;
  const float* wp = We + (size_t)d3 * H * 64 + h * 64;
  const float* ap = ae + h * 64;
  float s = 0.0f;
  for (int c = 0; c < 64; ++c) s += wp[c] * ap[c];
  wr[d3 * H + h] = s;
}

__global__ void k_alpha(const int* __restrict__ src, const int* __restrict__ dst,
                        const float* __restrict__ eattr, const float* __restrict__ meane,
                        const float* __restrict__ asrc, const float* __restrict__ adst,
                        const float* __restrict__ wr, float* __restrict__ alpha,
                        unsigned* __restrict__ amaxk, int H) {
  long tid = (long)blockIdx.x * blockDim.x + threadIdx.x;
  long tot = (long)(EE + NN) * H;
  if (tid >= tot) return;
  int e2 = (int)(tid / H), h = (int)(tid - (long)e2 * H);
  int s, d;
  const float* ea;
  if (e2 < EE) { s = src[e2]; d = dst[e2]; ea = eattr + (size_t)e2 * 3; }
  else { s = d = e2 - EE; ea = meane + (size_t)s * 3; }
  float av = ea[0] * wr[h] + ea[1] * wr[H + h] + ea[2] * wr[2 * H + h];
  float al = asrc[(size_t)s * H + h] + adst[(size_t)d * H + h] + av;
  al = (al > 0.0f) ? al : 0.2f * al;               // leaky_relu(0.2)
  alpha[tid] = al;
  atomicMax(amaxk + (size_t)d * H + h, f2key(al)); // segment max
}

__global__ void k_expd(const int* __restrict__ dst, float* __restrict__ alpha,
                       const unsigned* __restrict__ amaxk, float* __restrict__ denom,
                       int H) {
  long tid = (long)blockIdx.x * blockDim.x + threadIdx.x;
  long tot = (long)(EE + NN) * H;
  if (tid >= tot) return;
  int e2 = (int)(tid / H), h = (int)(tid - (long)e2 * H);
  int d = (e2 < EE) ? dst[e2] : (e2 - EE);
  float ex = __expf(alpha[tid] - key2f(amaxk[(size_t)d * H + h]));
  alpha[tid] = ex;
  atomicAdd(denom + (size_t)d * H + h, ex);
}

__global__ void k_scatter(const int* __restrict__ src, const int* __restrict__ dst,
                          const float* __restrict__ xl, const float* __restrict__ ex,
                          const float* __restrict__ denom, float* __restrict__ out,
                          int HC) {
  long tid = (long)blockIdx.x * blockDim.x + threadIdx.x;
  long tot = (long)(EE + NN) * HC;
  if (tid >= tot) return;
  int e2 = (int)(tid / HC);
  int r = (int)(tid - (long)e2 * HC);
  int h = r >> 6;
  int H = HC >> 6;
  int s, d;
  if (e2 < EE) { s = src[e2]; d = dst[e2]; } else { s = d = e2 - EE; }
  float attn = ex[(size_t)e2 * H + h] / (denom[(size_t)d * H + h] + 1e-16f);
  atomicAdd(out + (size_t)d * HC + r, xl[(size_t)s * HC + r] * attn);
}

__global__ void k_finish(float* __restrict__ out, const float* __restrict__ bias, int HC) {
  long tid = (long)blockIdx.x * blockDim.x + threadIdx.x;
  if (tid >= (long)NN * HC) return;
  float v = out[tid] + bias[tid % HC];
  out[tid] = (v > 0.0f) ? v : (__expf(v) - 1.0f);   // elu
}

// ---------------- edge MLP ----------------

// ef1[E x 64] = relu(concat(h[src], h[dst])[E x 128] @ mw1 + mb1).
// One wave per 16-edge tile; gather fused into the A fragment; 4 col-tiles per wave;
// software-pipelined like k_gemm.
__global__ __launch_bounds__(32) void k_mlp1(const float* __restrict__ hfin,
    const int* __restrict__ src, const int* __restrict__ dst,
    const _Float16* __restrict__ PB, const float* __restrict__ bias,
    float* __restrict__ Cout) {
  const int lane = threadIdx.x & 31;
  const int half = lane >> 4;
  const int l = lane & 15;
  const int m0 = blockIdx.x << 4;
  const int e = m0 + l;
  const float* hs = hfin + (size_t)src[e] * 64;
  const float* hd = hfin + (size_t)dst[e] * 64;
  const int klo_off = half ? 8 : 0;
  const int khi_off = half ? 24 : 16;
  const _Float16* bbase = PB + (lane << 4);

  v8f acc[4] = {{}, {}, {}, {}};

  // prologue: chunk 0 (all K-groups of 8 stay on one side of the 64 boundary)
  float4 p0 = *(const float4*)(hs + klo_off);
  float4 p1 = *(const float4*)(hs + klo_off + 4);
  float4 p2 = *(const float4*)(hs + khi_off);
  float4 p3 = *(const float4*)(hs + khi_off + 4);
  v16h c0 = *(const v16h*)(bbase);
  v16h c1 = *(const v16h*)(bbase + 512);
  v16h c2 = *(const v16h*)(bbase + 1024);
  v16h c3 = *(const v16h*)(bbase + 1536);
  v16h af = pack16(p0, p1, p2, p3);

#pragma unroll
  for (int k0 = 0; k0 < 128; k0 += 32) {
    const int kn = (k0 + 32 < 128) ? (k0 + 32) : k0;
    const int klo = kn + klo_off;
    const int khi = kn + khi_off;
    const float* plo = (klo < 64) ? (hs + klo) : (hd + klo - 64);
    const float* phi = (khi < 64) ? (hs + khi) : (hd + khi - 64);
    const _Float16* bn = bbase + (size_t)(kn >> 5) * 2048;
    float4 q0 = *(const float4*)plo;
    float4 q1 = *(const float4*)(plo + 4);
    float4 q2 = *(const float4*)phi;
    float4 q3 = *(const float4*)(phi + 4);
    v16h d0 = *(const v16h*)(bn);
    v16h d1 = *(const v16h*)(bn + 512);
    v16h d2 = *(const v16h*)(bn + 1024);
    v16h d3 = *(const v16h*)(bn + 1536);
    acc[0] = __builtin_amdgcn_wmma_f32_16x16x32_f16(false, af, false, c0, (short)0, acc[0], false, false);
    acc[1] = __builtin_amdgcn_wmma_f32_16x16x32_f16(false, af, false, c1, (short)0, acc[1], false, false);
    acc[2] = __builtin_amdgcn_wmma_f32_16x16x32_f16(false, af, false, c2, (short)0, acc[2], false, false);
    acc[3] = __builtin_amdgcn_wmma_f32_16x16x32_f16(false, af, false, c3, (short)0, acc[3], false, false);
    af = pack16(q0, q1, q2, q3);
    c0 = d0; c1 = d1; c2 = d2; c3 = d3;
  }

#pragma unroll
  for (int t = 0; t < 4; ++t) {
    const int col = (t << 4) + l;
    float bb = bias[col];
#pragma unroll
    for (int r = 0; r < 8; ++r) {
      float v = fmaxf(acc[t][r] + bb, 0.0f);
      Cout[(size_t)(m0 + r + (half << 3)) * 64 + col] = v;
    }
  }
}

__global__ void k_mlp2(const float* __restrict__ ef1, const float* __restrict__ mw2,
                       const float* __restrict__ mb2, float* __restrict__ out) {
  int tid = blockIdx.x * blockDim.x + threadIdx.x;
  if (tid >= EE * 3) return;
  int e = tid / 3, j = tid - e * 3;
  const float* f = ef1 + (size_t)e * 64;
  float s = mb2[j];
#pragma unroll
  for (int k = 0; k < 64; ++k) s += f[k] * mw2[k * 3 + j];
  out[tid] = s;
}

// ---------------- launch ----------------

extern "C" void kernel_launch(void* const* d_in, const int* in_sizes, int n_in,
                              void* d_out, int out_size, void* d_ws, size_t ws_size,
                              hipStream_t stream) {
  (void)in_sizes; (void)n_in; (void)out_size; (void)ws_size;
  const float* x      = (const float*)d_in[0];
  const int*   ei     = (const int*)d_in[1];
  const float* eattr  = (const float*)d_in[2];
  const float* enc_w1 = (const float*)d_in[3];
  const float* enc_b1 = (const float*)d_in[4];
  const float* enc_w2 = (const float*)d_in[5];
  const float* enc_b2 = (const float*)d_in[6];
  const float* Wl[3]  = {(const float*)d_in[7],  (const float*)d_in[13], (const float*)d_in[19]};
  const float* Wel[3] = {(const float*)d_in[8],  (const float*)d_in[14], (const float*)d_in[20]};
  const float* asl[3] = {(const float*)d_in[9],  (const float*)d_in[15], (const float*)d_in[21]};
  const float* adl[3] = {(const float*)d_in[10], (const float*)d_in[16], (const float*)d_in[22]};
  const float* ael[3] = {(const float*)d_in[11], (const float*)d_in[17], (const float*)d_in[23]};
  const float* bl[3]  = {(const float*)d_in[12], (const float*)d_in[18], (const float*)d_in[24]};
  const float* mw1 = (const float*)d_in[25];
  const float* mb1 = (const float*)d_in[26];
  const float* mw2 = (const float*)d_in[27];
  const float* mb2 = (const float*)d_in[28];
  const int* src = ei;
  const int* dst = ei + EE;
  float* out = (float*)d_out;

  char* wsb = (char*)d_ws;
  size_t off = 0;
  auto take = [&](size_t bytes) -> char* {
    char* p = wsb + off;
    off = (off + bytes + 255) & ~(size_t)255;
    return p;
  };
  float*    bufA  = (float*)take((size_t)NN * 512 * 4);
  float*    bufB  = (float*)take((size_t)NN * 512 * 4);
  float*    xlb   = (float*)take((size_t)NN * 512 * 4);
  float*    alpha = (float*)take((size_t)(EE + NN) * 8 * 4);
  float*    asrc  = (float*)take((size_t)NN * 8 * 4);
  float*    adst  = (float*)take((size_t)NN * 8 * 4);
  unsigned* amaxk = (unsigned*)take((size_t)NN * 8 * 4);
  float*    denom = (float*)take((size_t)NN * 8 * 4);
  float*    deg   = (float*)take((size_t)NN * 4);
  float*    macc  = (float*)take((size_t)NN * 3 * 4);
  float*    meane = (float*)take((size_t)NN * 3 * 4);
  float*    wered = (float*)take(24 * 4);
  float*    ef1   = (float*)take((size_t)EE * 64 * 4);
  _Float16* pb_enc = (_Float16*)take((size_t)128 * 64 * 2);
  _Float16* pb_w0  = (_Float16*)take((size_t)64 * 512 * 2);
  _Float16* pb_w1  = (_Float16*)take((size_t)512 * 512 * 2);
  _Float16* pb_w2  = (_Float16*)take((size_t)512 * 64 * 2);
  _Float16* pb_m1  = (_Float16*)take((size_t)128 * 64 * 2);
  const _Float16* PBl[3] = {pb_w0, pb_w1, pb_w2};

  auto cdiv = [](long a, long b) { return (int)((a + b - 1) / b); };

  // Pack weights to f16 fragment order
  k_pack_b<<<cdiv(128 * 64, 256), 256, 0, stream>>>(enc_w2, pb_enc, 128, 64);
  k_pack_b<<<cdiv(64 * 512, 256), 256, 0, stream>>>(Wl[0], pb_w0, 64, 512);
  k_pack_b<<<cdiv(512 * 512, 256), 256, 0, stream>>>(Wl[1], pb_w1, 512, 512);
  k_pack_b<<<cdiv(512 * 64, 256), 256, 0, stream>>>(Wl[2], pb_w2, 512, 64);
  k_pack_b<<<cdiv(128 * 64, 256), 256, 0, stream>>>(mw1, pb_m1, 128, 64);

  // Self-loop fill: per-dst mean edge attr
  k_zero<<<cdiv(NN, 256), 256, 0, stream>>>(deg, NN);
  k_zero<<<cdiv(NN * 3, 256), 256, 0, stream>>>(macc, NN * 3);
  k_deg<<<cdiv(EE, 256), 256, 0, stream>>>(dst, eattr, deg, macc);
  k_meandiv<<<cdiv(NN * 3, 256), 256, 0, stream>>>(macc, deg, meane);

  // Encoder: h = relu(x@w1+b1)@w2 + b2  (henc aliases xl scratch)
  float* henc = xlb;
  k_enc1<<<cdiv((long)NN * 128, 256), 256, 0, stream>>>(x, enc_w1, enc_b1, henc);
  k_gemm<<<dim3(NN / 16, 1), 32, 0, stream>>>(henc, pb_enc, enc_b2, bufA, NN, 128, 64, 0);

  // GAT layers
  int KinA[3] = {64, 512, 512};
  int Hh[3] = {8, 8, 1};
  float* hin = bufA;
  float* hout = bufB;
  for (int l = 0; l < 3; ++l) {
    int H = Hh[l], HC = H * 64;
    long ne = (long)(EE + NN) * H;
    long ns = (long)(EE + NN) * HC;
    k_gemm<<<dim3(NN / 16, HC / 64), 32, 0, stream>>>(hin, PBl[l], nullptr, xlb,
                                                      NN, KinA[l], HC, 0);
    k_attnode<<<cdiv((long)NN * H, 256), 256, 0, stream>>>(xlb, asl[l], adl[l], asrc, adst, H);
    k_wered<<<1, 32, 0, stream>>>(Wel[l], ael[l], wered, H);
    k_zero<<<cdiv((long)NN * H, 256), 256, 0, stream>>>((float*)amaxk, (long)NN * H);
    k_zero<<<cdiv((long)NN * H, 256), 256, 0, stream>>>(denom, (long)NN * H);
    k_alpha<<<cdiv(ne, 256), 256, 0, stream>>>(src, dst, eattr, meane, asrc, adst,
                                               wered, alpha, amaxk, H);
    k_expd<<<cdiv(ne, 256), 256, 0, stream>>>(dst, alpha, amaxk, denom, H);
    k_zero<<<cdiv((long)NN * HC, 256), 256, 0, stream>>>(hout, (long)NN * HC);
    k_scatter<<<cdiv(ns, 256), 256, 0, stream>>>(src, dst, xlb, alpha, denom, hout, HC);
    k_finish<<<cdiv((long)NN * HC, 256), 256, 0, stream>>>(hout, bl[l], HC);
    float* t = hin; hin = hout; hout = t;
  }

  // Edge MLP: relu(concat(h[src],h[dst]) @ mw1 + mb1) @ mw2 + mb2
  k_mlp1<<<EE / 16, 32, 0, stream>>>(hin, src, dst, pb_m1, mb1, ef1);
  k_mlp2<<<cdiv((long)EE * 3, 256), 256, 0, stream>>>(ef1, mw2, mb2, out);
}
